// cheb_conv_57200374448663
// MI455X (gfx1250) — matrix operand order repaired and checked by reading, not compiled, not run on hardware
//
#include <hip/hip_runtime.h>

// MI455X / gfx1250, wave32. Fused Chebyshev graph-conv:
//   out[b,m,o,t] = relu( sum_{k,f} ( sum_n x[b,n,f,t]*cheb[k,n,m] ) * theta[k,f,o] )
// Prep kernels convert x/cheb to bf16 in d_ws, pre-swizzled into WMMA operand
// layouts. Main kernel stages tiles with double-buffered
// global_load_async_to_lds_b128 (ASYNCcnt) and runs v_wmma_f32_16x16x32_bf16.

typedef __attribute__((ext_vector_type(16))) __bf16 v16bf;
typedef __attribute__((ext_vector_type(8)))  __bf16 v8bf;
typedef __attribute__((ext_vector_type(8)))  float  v8f;

#define B_    64
#define N_    1024
#define F_    16
#define T_    24
#define K_    3
#define O_    64
#define TQ    6     // t-columns per stage-1 pass (4 passes)
#define NSTEP 32    // n (K-dim) per WMMA step

#define XB_BYTES   ((size_t)B_ * 4 * 32 * TQ * NSTEP * 16 * 2)  // 50,331,648
#define CHEB_BYTES ((size_t)64 * 32 * K_ * NSTEP * 16 * 2)      //  6,291,456
#define XBLK  6144   // one (b,tq,nb) x tile, bytes
#define CBLK  3072   // one (mt,nb) cheb tile (all 3 k), bytes

// 16-bit A-matrix 16x32 lane/element -> K index (ISA 7.12.2)
__device__ __forceinline__ int kmap(int e, int lane) {
  return (e & 7) + ((e >> 3) << 4) + ((lane >> 4) << 3);
}

__device__ __forceinline__ v16bf cat16(v8bf lo, v8bf hi) {
  return __builtin_shufflevector(lo, hi, 0, 1, 2, 3, 4, 5, 6, 7, 8, 9, 10, 11,
                                 12, 13, 14, 15);
}
__device__ __forceinline__ v16bf ld16(const __bf16* p) {
  const v8bf lo = *(const v8bf*)p;
  const v8bf hi = *(const v8bf*)(p + 8);
  return cat16(lo, hi);
}
__device__ __forceinline__ unsigned pack2(float a, float b) {
  union { __bf16 h[2]; unsigned u; } r;
  r.h[0] = (__bf16)a;
  r.h[1] = (__bf16)b;
  return r.u;
}

// ---------- async global->LDS copy (CDNA5 data mover, ASYNCcnt) ----------
template <int OFF>
__device__ __forceinline__ void acp16(unsigned lds, unsigned voff,
                                      const void* sbase) {
  // per-lane 16B; INST_OFFSET applies to both LDS and global address
  asm volatile("global_load_async_to_lds_b128 %0, %1, %2 offset:%3" ::"v"(lds),
               "v"(voff), "s"(sbase), "i"(OFF));
}
__device__ __forceinline__ void acp_x(unsigned lds, unsigned voff,
                                      const void* s) {  // 6144 B = 12 x 512
  acp16<0>(lds, voff, s);    acp16<512>(lds, voff, s);
  acp16<1024>(lds, voff, s); acp16<1536>(lds, voff, s);
  acp16<2048>(lds, voff, s); acp16<2560>(lds, voff, s);
  acp16<3072>(lds, voff, s); acp16<3584>(lds, voff, s);
  acp16<4096>(lds, voff, s); acp16<4608>(lds, voff, s);
  acp16<5120>(lds, voff, s); acp16<5632>(lds, voff, s);
}
__device__ __forceinline__ void acp_c(unsigned lds, unsigned voff,
                                      const void* s) {  // 3072 B = 6 x 512
  acp16<0>(lds, voff, s);    acp16<512>(lds, voff, s);
  acp16<1024>(lds, voff, s); acp16<1536>(lds, voff, s);
  acp16<2048>(lds, voff, s); acp16<2560>(lds, voff, s);
}

// =============== prep: x -> bf16, B1-operand tile layout ===============
// xB[((b*4+tq)*32+nb)] block: [t][lane (f = lane&15)][elem (n via kmap)], 6144 B contiguous
__global__ __launch_bounds__(256) void prep_x(const float* __restrict__ x,
                                              __bf16* __restrict__ xB) {
  const int idx = blockIdx.x * 256 + threadIdx.x;  // 1,572,864 total
  const int L   = idx & 31;
  const int t   = (idx >> 5) % TQ;
  const int bl  = idx / (TQ * 32);  // (b*4+tq)*32+nb
  const int nb  = bl & 31;
  const int tq  = (bl >> 5) & 3;
  const int b   = bl >> 7;
  const float* src =
      x + (((size_t)b * N_ + nb * 32) * F_ + (L & 15)) * T_ + tq * TQ + t;
  __bf16* dst = xB + (size_t)idx * 16;
#pragma unroll
  for (int e = 0; e < 16; ++e)
    dst[e] = (__bf16)src[(size_t)kmap(e, L) * F_ * T_];
}

// =============== prep: cheb -> bf16, A1-operand swizzle ===============
// chebA[(mt*32+nb)] block: [k][opLane][elem], 3072 B contiguous
__global__ __launch_bounds__(256) void prep_cheb(const float* __restrict__ cheb,
                                                 __bf16* __restrict__ chebA) {
  const int idx = blockIdx.x * 256 + threadIdx.x;  // 196,608 total
  const int L   = idx & 31;
  const int k   = (idx >> 5) % K_;
  const int q   = idx / (32 * K_);
  const int nb  = q & 31;
  const int mt  = q >> 5;
  __bf16* dst = chebA + (size_t)idx * 16;
  const int m  = mt * 16 + (L & 15);
#pragma unroll
  for (int e = 0; e < 16; ++e)
    dst[e] = (__bf16)cheb[((size_t)k * N_ + nb * 32 + kmap(e, L)) * N_ + m];
}

// ============================ main kernel ============================
__global__ __launch_bounds__(32)
void cheb_conv_fused_wmma(const __bf16* __restrict__ xB,
                          const __bf16* __restrict__ chebA,
                          const float* __restrict__ theta,
                          float* __restrict__ out) {
  // LDS: 2*6144 + 2*3072 + 8192 + 36864 = 63,488 bytes
  __shared__ __attribute__((aligned(16))) __bf16 lds_x[2][TQ][NSTEP][16];
  __shared__ __attribute__((aligned(16))) __bf16 lds_chebA[2][K_][NSTEP][16];
  __shared__ __attribute__((aligned(16))) __bf16 thetaA[4][2][NSTEP][16];
  __shared__ __attribute__((aligned(16))) __bf16 yB[16][48][T_];  // [m][K'][t]

  const int lane = threadIdx.x;  // one wave per WG
  const int hb   = lane >> 4;
  const int mr   = lane & 15;
  const int b    = blockIdx.x >> 6;
  const int mt   = blockIdx.x & 63;
  const int m0   = mt << 4;

  const v8f  vz8  = {0.f, 0.f, 0.f, 0.f, 0.f, 0.f, 0.f, 0.f};
  const v8bf vzb8 = {(__bf16)0.f, (__bf16)0.f, (__bf16)0.f, (__bf16)0.f,
                     (__bf16)0.f, (__bf16)0.f, (__bf16)0.f, (__bf16)0.f};
  const v16bf vzb16 = cat16(vzb8, vzb8);

  // ---- stage theta (A2-operand swizzle, K' 48..63 zero) ----
  for (int idx = lane; idx < 4 * 2 * NSTEP * 16; idx += 32) {
    const int e  = idx & 15;
    const int L  = (idx >> 4) & 31;
    const int ks = (idx >> 9) & 1;
    const int ot = idx >> 10;
    const int Kp = kmap(e, L) + 32 * ks;
    const int o  = ot * 16 + (L & 15);
    thetaA[ot][ks][L][e] = (Kp < 48) ? (__bf16)theta[Kp * 64 + o] : (__bf16)0.0f;
  }
  __syncthreads();

  // per-lane LDS destination addresses (low 32 bits of generic ptr = LDS off)
  const unsigned voff = (unsigned)lane * 16u;
  const unsigned lx0 = (unsigned)(size_t)&lds_x[0][0][0][0] + voff;
  const unsigned lx1 = (unsigned)(size_t)&lds_x[1][0][0][0] + voff;
  const unsigned lc0 = (unsigned)(size_t)&lds_chebA[0][0][0][0] + voff;
  const unsigned lc1 = (unsigned)(size_t)&lds_chebA[1][0][0][0] + voff;

  const char* xbase = (const char*)xB + (size_t)b * 128 * XBLK;
  const char* cbase = (const char*)chebA + (size_t)mt * 32 * CBLK;

  // issue one double-buffered batch: 12 + 6 async b128 bursts (18 ops)
  auto issue = [&](int ib, int buf) {
    acp_x(buf ? lx1 : lx0, voff, xbase + (size_t)ib * XBLK);
    acp_c(buf ? lc1 : lc0, voff, cbase + (size_t)(ib & 31) * CBLK);
  };

  v8f acc[K_][TQ];
#pragma unroll
  for (int k = 0; k < K_; ++k)
#pragma unroll
    for (int t = 0; t < TQ; ++t) acc[k][t] = vz8;

  issue(0, 0);
  int buf = 0;
  for (int ib = 0; ib < 128; ++ib) {  // ib = tq*32 + nb
    if (ib + 1 < 128) {
      issue(ib + 1, buf ^ 1);
      asm volatile("s_wait_asynccnt 18" ::: "memory");  // batch ib landed
    } else {
      asm volatile("s_wait_asynccnt 0" ::: "memory");
    }
    // ---- 18 WMMAs on buffer `buf`
    v16bf a[K_];
#pragma unroll
    for (int k = 0; k < K_; ++k) a[k] = ld16(&lds_chebA[buf][k][lane][0]);
#pragma unroll
    for (int t = 0; t < TQ; ++t) {
      const v16bf bb = ld16(&lds_x[buf][t][lane][0]);
#pragma unroll
      for (int k = 0; k < K_; ++k)
        acc[k][t] = __builtin_amdgcn_wmma_f32_16x16x32_bf16(
            false, a[k], false, bb, (short)0, acc[k][t], false, false);
    }
    // ---- end of a tq pass: commit y (packed b32 stores) and reset acc
    if ((ib & 31) == 31) {
      const int tq = ib >> 5;
#pragma unroll
      for (int k = 0; k < K_; ++k)
#pragma unroll
        for (int v = 0; v < 8; ++v) {
          unsigned* dst =
              (unsigned*)&yB[v + 8 * hb][k * 16 + mr][tq * TQ];
#pragma unroll
          for (int tp = 0; tp < TQ / 2; ++tp)
            dst[tp] = pack2(acc[k][2 * tp][v], acc[k][2 * tp + 1][v]);
        }
#pragma unroll
      for (int k = 0; k < K_; ++k)
#pragma unroll
        for (int t = 0; t < TQ; ++t) acc[k][t] = vz8;
    }
    buf ^= 1;
  }
  __syncthreads();

  // =========================== STAGE 2 ===========================
  // out[b][m0+m][o][t] = relu( sum_{K'} thetaA . yB ), D = [o16 x t16]
  v16bf a2[4][2];
#pragma unroll
  for (int ot = 0; ot < 4; ++ot)
#pragma unroll
    for (int ks = 0; ks < 2; ++ks) a2[ot][ks] = ld16(&thetaA[ot][ks][lane][0]);

  for (int m = 0; m < 16; ++m) {
#pragma unroll
    for (int tt = 0; tt < 2; ++tt) {
      v16bf b0, b1;
      {
        const int tc = tt * 16 + mr;
        const int tr = (tc < 24) ? tc : 23;
#pragma unroll
        for (int e = 0; e < 16; ++e) {
          const int Kp0 = kmap(e, lane);
          const int Kp1 = Kp0 + 32;
          b0[e] = yB[m][Kp0][tr];
          b1[e] = (Kp1 < 48) ? yB[m][Kp1][tr] : (__bf16)0.0f;
        }
      }
#pragma unroll
      for (int ot = 0; ot < 4; ++ot) {
        v8f c = vz8;
        c = __builtin_amdgcn_wmma_f32_16x16x32_bf16(false, a2[ot][0], false, b0,
                                                    (short)0, c, false, false);
        c = __builtin_amdgcn_wmma_f32_16x16x32_bf16(false, a2[ot][1], false, b1,
                                                    (short)0, c, false, false);
        const int tcol = tt * 16 + mr;
        if (tcol < 24) {
          const size_t base = (((size_t)b * N_ + m0 + m) * O_) * T_ + tcol;
#pragma unroll
          for (int v = 0; v < 8; ++v) {
            const int o = ot * 16 + v + 8 * hb;
            float r = c[v];
            r = (r > 0.f) ? r : 0.f;
            out[base + (size_t)o * T_] = r;
          }
        }
      }
    }
  }
}

extern "C" void kernel_launch(void* const* d_in, const int* in_sizes, int n_in,
                              void* d_out, int out_size, void* d_ws,
                              size_t ws_size, hipStream_t stream) {
  const float* x     = (const float*)d_in[0];
  const float* cheb  = (const float*)d_in[1];
  const float* theta = (const float*)d_in[2];
  float* out = (float*)d_out;

  __bf16* xB    = (__bf16*)d_ws;
  __bf16* chebA = (__bf16*)((char*)d_ws + XB_BYTES);

  prep_x<<<dim3(1572864 / 256), dim3(256), 0, stream>>>(x, xB);
  prep_cheb<<<dim3(196608 / 256), dim3(256), 0, stream>>>(cheb, chebA);

  dim3 grid(B_ * (N_ / 16));  // 4096 single-wave WGs
  cheb_conv_fused_wmma<<<grid, dim3(32), 0, stream>>>(xB, chebA, theta, out);
}
